// QuantumFeedForward_65481071403869
// MI455X (gfx1250) — matrix operand order repaired and checked by reading, not compiled, not run on hardware
//
#include <hip/hip_runtime.h>
#include <math.h>

// Problem constants (from reference)
#define N_TOK   16384      // B*S = 8*2048
#define EMBED   512
#define FFN     2048
#define NQ      10
#define HSTRIDE 2056       // FFN + 8 halfs pad -> 4112B row stride, bank-conflict-free b128

typedef __attribute__((ext_vector_type(16))) _Float16 v16h;
typedef __attribute__((ext_vector_type(8)))  float    v8f;

union Frag { v16h v; uint4 q[2]; };

// XOR-support sets of the depth-2 ring-CNOT chain (bit j set => qubit j in product)
__constant__ int QMASK[NQ] = {683, 1021, 1018, 1013, 1002, 981, 938, 853, 682, 341};

// ---------------- W2 (EMBED x FFN) fp32 -> fp16 into workspace ----------------
__global__ __launch_bounds__(256)
void w2_to_f16_kernel(const float* __restrict__ w2, _Float16* __restrict__ w2h) {
  int i = (blockIdx.x * 256 + threadIdx.x) * 4;       // EMBED*FFN divisible by 4*256*grid
  float4 v = *(const float4*)(w2 + i);
  union { _Float16 h[4]; uint2 u; } p;
  p.h[0] = (_Float16)v.x; p.h[1] = (_Float16)v.y;
  p.h[2] = (_Float16)v.z; p.h[3] = (_Float16)v.w;
  *(uint2*)(w2h + i) = p.u;
}

// ---------------- fused quantum-collapse + FFN kernel ----------------
// block = 256 threads (8 wave32), one block per 16-token strip
__global__ __launch_bounds__(256)
void qffn_kernel(const float* __restrict__ x,
                 const float* __restrict__ ry_theta,
                 const float* __restrict__ W1,
                 const float* __restrict__ b1,
                 const _Float16* __restrict__ w2h,
                 const float* __restrict__ b2,
                 float* __restrict__ out) {
  extern __shared__ char smem[];
  _Float16* hT = (_Float16*)smem;                              // 16 x HSTRIDE f16
  float*    qS = (float*)(smem + 16 * HSTRIDE * sizeof(_Float16)); // 16 x 16 f32

  const int tid   = threadIdx.x;
  const int tBase = blockIdx.x * 16;

  // ---- stage 1: q[t][w] = prod_{j in S_w} cos(theta_j) * cos(x[t,j])
  if (tid < 16) {
    const float* xr = x + (size_t)(tBase + tid) * EMBED;
    float m[NQ];
#pragma unroll
    for (int j = 0; j < NQ; ++j)
      m[j] = cosf(ry_theta[j]) * cosf(xr[j]);
#pragma unroll
    for (int w = 0; w < NQ; ++w) {
      float p = 1.0f;
      int msk = QMASK[w];
#pragma unroll
      for (int j = 0; j < NQ; ++j)
        if ((msk >> j) & 1) p *= m[j];
      qS[tid * 16 + w] = p;
    }
  }
  __syncthreads();

  // ---- stage 2: h strip (16 x FFN) = relu(q @ W1^T + b1), f16 in LDS
  for (int k = tid; k < FFN; k += 256) {
    float w1r[NQ];
#pragma unroll
    for (int j = 0; j < NQ; ++j) w1r[j] = W1[k * NQ + j];
    float bb = b1[k];
#pragma unroll
    for (int t = 0; t < 16; ++t) {
      float s = bb;
#pragma unroll
      for (int j = 0; j < NQ; ++j) s += qS[t * 16 + j] * w1r[j];
      hT[t * HSTRIDE + k] = (_Float16)(s > 0.0f ? s : 0.0f);
    }
  }
  __syncthreads();

  // ---- stage 3: out strip (16 x EMBED) = h @ W2^T + b2 via WMMA f16->f32
  const int lane = tid & 31;
  const int wave = tid >> 5;         // 8 waves x 4 tiles x 16 cols = EMBED
  const int n16  = lane & 15;        // N (and A-row M) index within tile
  const int hi   = lane >> 4;        // lane half selects K sub-block

  // A fragment source: row m = n16; lanes>=16 start at K+8 (ISA 16-bit A layout)
  const _Float16* ap = hT + n16 * HSTRIDE + hi * 8;

  // B fragment sources: 4 rows of W2 (one per N-tile); lanes>=16 start at K+8
  const _Float16* bp[4];
#pragma unroll
  for (int nt = 0; nt < 4; ++nt) {
    int e = wave * 64 + nt * 16 + n16;
    bp[nt] = w2h + (size_t)e * FFN + hi * 8;
  }

  v8f acc[4] = {};

#pragma unroll 2
  for (int k0 = 0; k0 < FFN; k0 += 32) {
    Frag a;
    a.q[0] = *(const uint4*)(ap + k0);        // K = k0 + hi*8 + [0..7]
    a.q[1] = *(const uint4*)(ap + k0 + 16);   // K = k0 + hi*8 + 16 + [0..7]
#pragma unroll
    for (int nt = 0; nt < 4; ++nt) {
      Frag b;
      b.q[0] = *(const uint4*)(bp[nt] + k0);      // K = k0 + hi*8 + [0..7]
      b.q[1] = *(const uint4*)(bp[nt] + k0 + 16); // K = k0 + hi*8 + 16 + [0..7]
      acc[nt] = __builtin_amdgcn_wmma_f32_16x16x32_f16(
          /*neg_a=*/false, a.v, /*neg_b=*/false, b.v,
          /*c_mod=*/(short)0, acc[nt], /*reuse_a=*/false, /*reuse_b=*/false);
    }
  }

  // ---- epilogue: add b2, store per ISA 16x16 f32 C/D layout
#pragma unroll
  for (int nt = 0; nt < 4; ++nt) {
    int e = wave * 64 + nt * 16 + n16;
    float bb = b2[e];
    float* op = out + (size_t)(tBase + hi * 8) * EMBED + e;
#pragma unroll
    for (int r = 0; r < 8; ++r)
      op[(size_t)r * EMBED] = acc[nt][r] + bb;   // M = r + 8*hi, N = n16
  }
}

extern "C" void kernel_launch(void* const* d_in, const int* in_sizes, int n_in,
                              void* d_out, int out_size, void* d_ws, size_t ws_size,
                              hipStream_t stream) {
  const float* x  = (const float*)d_in[0];
  const float* th = (const float*)d_in[1];
  const float* W1 = (const float*)d_in[2];
  const float* b1 = (const float*)d_in[3];
  const float* W2 = (const float*)d_in[4];
  const float* b2 = (const float*)d_in[5];
  float* out = (float*)d_out;

  _Float16* w2h = (_Float16*)d_ws;   // 512*2048*2 = 2 MB of workspace

  // W2 -> f16 (1,048,576 elems, 4 per thread)
  w2_to_f16_kernel<<<dim3((EMBED * FFN) / (256 * 4)), dim3(256), 0, stream>>>(W2, w2h);

  // fused kernel: one block per 16-token strip
  const size_t smem_bytes = 16 * HSTRIDE * sizeof(_Float16) + 16 * 16 * sizeof(float);
  qffn_kernel<<<dim3(N_TOK / 16), dim3(256), smem_bytes, stream>>>(
      x, th, W1, b1, w2h, b2, out);
}
